// Emb_ATT_Layers_22110491640676
// MI455X (gfx1250) — matrix operand order repaired and checked by reading, not compiled, not run on hardware
//
#include <hip/hip_runtime.h>
#include <hip/hip_bf16.h>
#include <cstddef>

#define N_NODES 50000
#define N_REL   32
#define N_EDGES 1600000
#define EMB     64
#define HID     64
#define LBL     8

#define BST 72   // f16 Bt row stride (halves)  -> conflict-free bank pattern
#define FST 65   // f32 staging row stride (floats)
#define CST 72   // f16 ctx row stride (halves)

typedef __attribute__((ext_vector_type(16))) _Float16 v16h;
typedef __attribute__((ext_vector_type(8)))  float    v8f;

// ---------------------------------------------------------------------------
// WMMA helpers (CDNA5 wave32, v_wmma_f32_16x16x32_f16)
// A 16x32 f16: lane m=lane&15 holds row m; c=lane>>4 selects K-halves:
//   elems 0..7  = A[m][k0 + c*8 + e],  elems 8..15 = A[m][k0 + 16 + c*8 + e]
// B 32x16 f16: lane holds column n=lane&15, same K layout as A (rows k0 + c*8 + e, k0 + 16 + c*8 + e)
// C 16x16 f32: VGPR j: lanes0-15 -> (row j, col lane); lanes16-31 -> (row j+8)
// ---------------------------------------------------------------------------
__device__ __forceinline__ v8f wmma32(v16h a, v16h b, v8f c) {
  return __builtin_amdgcn_wmma_f32_16x16x32_f16(false, a, false, b, (short)0, c,
                                                false, false);
}

__device__ __forceinline__ v16h make_a_f32(const float* row, int k0, int c, float s) {
  const float* p0 = row + k0 + c * 8;
  const float* p1 = row + k0 + 16 + c * 8;
  v16h a;
#pragma unroll
  for (int e = 0; e < 8; ++e) {
    a[e]     = (_Float16)(p0[e] * s);
    a[8 + e] = (_Float16)(p1[e] * s);
  }
  return a;
}

__device__ __forceinline__ v16h make_a_h(const _Float16* as, int m, int k0, int c,
                                         int stride) {
  const _Float16* p0 = as + m * stride + k0 + c * 8;
  const _Float16* p1 = p0 + 16;
  v16h a;
#pragma unroll
  for (int e = 0; e < 8; ++e) { a[e] = p0[e]; a[8 + e] = p1[e]; }
  return a;
}

// B fragment from f16 LDS Bt [outDim][BST]
__device__ __forceinline__ v16h make_b_h(const _Float16* wt, int n_glob, int k0, int c) {
  const _Float16* p = wt + n_glob * BST + k0 + c * 8;
  v16h b;
#pragma unroll
  for (int e = 0; e < 8; ++e) { b[e] = p[e]; b[8 + e] = p[16 + e]; }
  return b;
}

// ---------------------------------------------------------------------------
// Stage 1: MHA, keep only sequence position 0.  4 waves/block, 16 nodes/wave.
// Weights staged once per block into LDS as f16 Bt tiles.
// ---------------------------------------------------------------------------
#define MHA_WAVES 4
#define MHA_WEIGHT_BYTES ((192 + 64) * BST * 2)
#define MHA_WAVE_BYTES   ((16*FST + 16*FST + 4*16*FST) * 4 + 16*CST*2)

__global__ void mha_kernel(const float* __restrict__ emb,
                           const float* __restrict__ inW,  const float* __restrict__ inB,
                           const float* __restrict__ outW, const float* __restrict__ outB,
                           float* __restrict__ x) {
  extern __shared__ char smem[];
  _Float16* wT = (_Float16*)smem;                 // [256 rows][BST] Bt (in||out)
  for (int idx = threadIdx.x; idx < 256 * 64; idx += blockDim.x) {
    const int row = idx >> 6, k = idx & 63;
    const float v = (row < 192) ? inW[row * 64 + k] : outW[(row - 192) * 64 + k];
    wT[row * BST + k] = (_Float16)v;
  }
  __syncthreads();

  const int lane = threadIdx.x & 31;
  const int wave = threadIdx.x >> 5;
  int base = (blockIdx.x * MHA_WAVES + wave) * 16;
  if (base > N_NODES - 16) base = N_NODES - 16;   // clamp: duplicates write same data

  char* wsm = smem + MHA_WEIGHT_BYTES + wave * MHA_WAVE_BYTES;
  float*    q_s   = (float*)wsm;                  // [16][FST]
  float*    k_s   = q_s + 16 * FST;               // [16][FST]
  float*    v_s   = k_s + 16 * FST;               // [4][16][FST]
  _Float16* ctx_s = (_Float16*)(v_s + 4 * 16 * FST);  // [16][CST]

  const int m = lane & 15, c = lane >> 4;
  const int node = base + m;

  float sc[2][4];
#pragma unroll
  for (int l = 0; l < 4; ++l) {
    const float* row = emb + ((size_t)l * N_NODES + node) * 64;
    v16h a0 = make_a_f32(row, 0, c, 1.f);
    v16h a1 = make_a_f32(row, 32, c, 1.f);
    const int t0 = (l == 0) ? 0 : 4;              // q tiles only for l==0
    for (int t = t0; t < 12; ++t) {
      const int ng = t * 16 + m;
      v8f C = {};
      C = wmma32(a0, make_b_h(wT, ng, 0, c), C);
      C = wmma32(a1, make_b_h(wT, ng, 32, c), C);
      const float bias = inB[ng];
      float* S = (t < 4) ? q_s : (t < 8 ? k_s : (v_s + l * 16 * FST));
      const int ct = (t < 4) ? t : (t < 8 ? t - 4 : t - 8);
#pragma unroll
      for (int j = 0; j < 8; ++j)
        S[(j + 8 * c) * FST + ct * 16 + m] = C[j] + bias;
    }
    __syncthreads();
#pragma unroll
    for (int p = 0; p < 2; ++p) {
      const int pid = lane + 32 * p;
      const int nn = pid >> 2, hh = pid & 3;
      float s = 0.f;
#pragma unroll
      for (int d = 0; d < 16; ++d)
        s += q_s[nn * FST + hh * 16 + d] * k_s[nn * FST + hh * 16 + d];
      sc[p][l] = s * 0.25f;                       // 1/sqrt(16)
    }
    __syncthreads();                              // k_s reused next l
  }

#pragma unroll
  for (int p = 0; p < 2; ++p) {
    const int pid = lane + 32 * p;
    const int nn = pid >> 2, hh = pid & 3;
    float mx = fmaxf(fmaxf(sc[p][0], sc[p][1]), fmaxf(sc[p][2], sc[p][3]));
    float w0 = __expf(sc[p][0] - mx), w1 = __expf(sc[p][1] - mx);
    float w2 = __expf(sc[p][2] - mx), w3 = __expf(sc[p][3] - mx);
    const float inv = 1.f / (w0 + w1 + w2 + w3);
#pragma unroll
    for (int d = 0; d < 16; ++d) {
      const int off = nn * FST + hh * 16 + d;
      float v = (w0 * v_s[off] + w1 * v_s[16 * FST + off] +
                 w2 * v_s[32 * FST + off] + w3 * v_s[48 * FST + off]) * inv;
      ctx_s[nn * CST + hh * 16 + d] = (_Float16)v;
    }
  }
  __syncthreads();

  v16h a0 = make_a_h(ctx_s, m, 0, c, CST);
  v16h a1 = make_a_h(ctx_s, m, 32, c, CST);
  const _Float16* oT = wT + 192 * BST;
  for (int t = 0; t < 4; ++t) {
    const int ng = t * 16 + m;
    v8f C = {};
    C = wmma32(a0, make_b_h(oT, ng, 0, c), C);
    C = wmma32(a1, make_b_h(oT, ng, 32, c), C);
    const float bias = outB[ng];
#pragma unroll
    for (int j = 0; j < 8; ++j) {
      const int rn = base + j + 8 * c;
      x[(size_t)rn * 64 + t * 16 + m] = C[j] + bias;
    }
  }
}

// ---------------------------------------------------------------------------
// Stage 2: per-(relation,dst) segment sum via f32 atomics. 4 lanes/edge,
// edge metadata loaded once and broadcast with wave32 shuffles.
// ---------------------------------------------------------------------------
__global__ void agg_kernel(const float* __restrict__ xbuf,
                           const int* __restrict__ ei, const int* __restrict__ et,
                           float* __restrict__ agg, float* __restrict__ cnt,
                           int do_cnt) {
  const long long t = (long long)blockIdx.x * blockDim.x + threadIdx.x;
  const int e = (int)(t >> 2);
  const int part = (int)(t & 3);
  const int lane = threadIdx.x & 31;
  int src = 0, dst = 0, r = 0;
  if (part == 0 && e < N_EDGES) {
    src = ei[e];
    dst = ei[N_EDGES + e];
    r   = et[e];
  }
  const int lsrc = lane & ~3;
  src = __shfl(src, lsrc);
  dst = __shfl(dst, lsrc);
  r   = __shfl(r, lsrc);
  if (e >= N_EDGES) return;

  const size_t seg = (size_t)r * N_NODES + dst;
  const float4* sp = (const float4*)(xbuf + (size_t)src * 64 + part * 16);
  float* ap = agg + seg * 64 + part * 16;
#pragma unroll
  for (int q = 0; q < 4; ++q) {
    float4 v = sp[q];
    unsafeAtomicAdd(ap + q * 4 + 0, v.x);
    unsafeAtomicAdd(ap + q * 4 + 1, v.y);
    unsafeAtomicAdd(ap + q * 4 + 2, v.z);
    unsafeAtomicAdd(ap + q * 4 + 3, v.w);
  }
  if (do_cnt && part == 0) unsafeAtomicAdd(cnt + seg, 1.f);
}

// ---------------------------------------------------------------------------
// Stage 3: RGCN layer 1.  W1 + root1 transposed in LDS as f16 (stride-padded,
// 33*64*72*2 = 297KB <= 320KB WGP LDS).  Wave owns a 16-node tile, loops all
// 33 "relations" accumulating 4 f32 C-tiles; no output atomics.
// ---------------------------------------------------------------------------
#define R1_WAVES 8
__global__ void rgcn1_kernel(const float* __restrict__ agg, const float* __restrict__ cnt,
                             const float* __restrict__ xbuf,
                             const float* __restrict__ w1, const float* __restrict__ root1,
                             const float* __restrict__ b1, float* __restrict__ h1) {
  extern __shared__ _Float16 wt[];              // [33][64][BST]
  for (int idx = threadIdx.x; idx < 33 * 4096; idx += blockDim.x) {
    const int r = idx >> 12, rem = idx & 4095;
    const int k = rem >> 6, n = rem & 63;
    const float v = (r < 32) ? w1[(size_t)r * 4096 + k * 64 + n] : root1[k * 64 + n];
    wt[(r * 64 + n) * BST + k] = (_Float16)v;
  }
  __syncthreads();

  const int lane = threadIdx.x & 31, wave = threadIdx.x >> 5;
  int base = (blockIdx.x * R1_WAVES + wave) * 16;
  if (base > N_NODES - 16) base = N_NODES - 16;
  const int m = lane & 15, c = lane >> 4;
  const int node = base + m;

  v8f C[4] = {};
  for (int r = 0; r < 33; ++r) {
    const float* rowp;
    float scale;
    if (r < 32) {
      const float ct = cnt[(size_t)r * N_NODES + node];
      scale = 1.f / fmaxf(ct, 1.f);
      rowp = agg + ((size_t)r * N_NODES + node) * 64;
      if (r + 1 < 32) {                         // prefetch next relation's row
        const float* nxt = agg + ((size_t)(r + 1) * N_NODES + node) * 64;
        __builtin_prefetch(nxt, 0, 1);
        __builtin_prefetch(nxt + 32, 0, 1);
      }
    } else {
      scale = 1.f;
      rowp = xbuf + (size_t)node * 64;
    }
    v16h a0 = make_a_f32(rowp, 0, c, scale);
    v16h a1 = make_a_f32(rowp, 32, c, scale);
    const _Float16* bt = wt + r * 64 * BST;
#pragma unroll
    for (int t = 0; t < 4; ++t) {
      C[t] = wmma32(a0, make_b_h(bt, t * 16 + m, 0, c), C[t]);
      C[t] = wmma32(a1, make_b_h(bt, t * 16 + m, 32, c), C[t]);
    }
  }
#pragma unroll
  for (int t = 0; t < 4; ++t) {
    const float bias = b1[t * 16 + m];
#pragma unroll
    for (int j = 0; j < 8; ++j) {
      const int rn = base + j + 8 * c;
      h1[(size_t)rn * 64 + t * 16 + m] = fmaxf(C[t][j] + bias, 0.f);
    }
  }
}

// ---------------------------------------------------------------------------
// Stage 4: RGCN layer 2 (64 -> 8, padded to 16 cols) + sigmoid.
// ---------------------------------------------------------------------------
__global__ void rgcn2_kernel(const float* __restrict__ agg, const float* __restrict__ cnt,
                             const float* __restrict__ h1,
                             const float* __restrict__ w2, const float* __restrict__ root2,
                             const float* __restrict__ b2, float* __restrict__ out) {
  extern __shared__ _Float16 wt2[];             // [33][16][BST]
  for (int idx = threadIdx.x; idx < 33 * 1024; idx += blockDim.x) {
    const int r = idx >> 10, rem = idx & 1023;
    const int n = rem >> 6, k = rem & 63;
    float v = 0.f;
    if (n < LBL) v = (r < 32) ? w2[(size_t)r * 512 + k * 8 + n] : root2[k * 8 + n];
    wt2[(r * 16 + n) * BST + k] = (_Float16)v;
  }
  __syncthreads();

  const int lane = threadIdx.x & 31, wave = threadIdx.x >> 5;
  int base = (blockIdx.x * R1_WAVES + wave) * 16;
  if (base > N_NODES - 16) base = N_NODES - 16;
  const int m = lane & 15, c = lane >> 4;
  const int node = base + m;

  v8f C = {};
  for (int r = 0; r < 33; ++r) {
    const float* rowp;
    float scale;
    if (r < 32) {
      const float ct = cnt[(size_t)r * N_NODES + node];
      scale = 1.f / fmaxf(ct, 1.f);
      rowp = agg + ((size_t)r * N_NODES + node) * 64;
      if (r + 1 < 32) {
        const float* nxt = agg + ((size_t)(r + 1) * N_NODES + node) * 64;
        __builtin_prefetch(nxt, 0, 1);
        __builtin_prefetch(nxt + 32, 0, 1);
      }
    } else {
      scale = 1.f;
      rowp = h1 + (size_t)node * 64;
    }
    v16h a0 = make_a_f32(rowp, 0, c, scale);
    v16h a1 = make_a_f32(rowp, 32, c, scale);
    const _Float16* bt = wt2 + r * 16 * BST;
    C = wmma32(a0, make_b_h(bt, m, 0, c), C);
    C = wmma32(a1, make_b_h(bt, m, 32, c), C);
  }
  if (m < LBL) {
    const float bias = b2[m];
#pragma unroll
    for (int j = 0; j < 8; ++j) {
      const int rn = base + j + 8 * c;
      const float v = C[j] + bias;
      out[(size_t)rn * LBL + m] = 1.f / (1.f + __expf(-v));
    }
  }
}

// ---------------------------------------------------------------------------
extern "C" void kernel_launch(void* const* d_in, const int* in_sizes, int n_in,
                              void* d_out, int out_size, void* d_ws, size_t ws_size,
                              hipStream_t stream) {
  (void)in_sizes; (void)n_in; (void)out_size; (void)ws_size;
  const float* emb   = (const float*)d_in[0];
  const int*   ei    = (const int*)d_in[1];
  const int*   et    = (const int*)d_in[2];
  const float* inW   = (const float*)d_in[3];
  const float* inB   = (const float*)d_in[4];
  const float* outW  = (const float*)d_in[5];
  const float* outB  = (const float*)d_in[6];
  const float* w1    = (const float*)d_in[7];
  const float* root1 = (const float*)d_in[8];
  const float* b1    = (const float*)d_in[9];
  const float* w2    = (const float*)d_in[10];
  const float* root2 = (const float*)d_in[11];
  const float* b2    = (const float*)d_in[12];
  float* out = (float*)d_out;

  float* x   = (float*)d_ws;                           // N*64
  float* h1  = x + (size_t)N_NODES * 64;               // N*64
  float* cnt = h1 + (size_t)N_NODES * 64;              // R*N
  float* agg = cnt + (size_t)N_REL * N_NODES;          // R*N*64
  const size_t aggBytes = (size_t)N_REL * N_NODES * 64 * sizeof(float);
  const size_t cntBytes = (size_t)N_REL * N_NODES * sizeof(float);

  hipMemsetAsync(agg, 0, aggBytes, stream);
  hipMemsetAsync(cnt, 0, cntBytes, stream);

  const int mhaBlocks = (N_NODES + MHA_WAVES * 16 - 1) / (MHA_WAVES * 16);
  const size_t mhaLds = MHA_WEIGHT_BYTES + MHA_WAVES * MHA_WAVE_BYTES;
  mha_kernel<<<mhaBlocks, MHA_WAVES * 32, mhaLds, stream>>>(
      emb, inW, inB, outW, outB, x);

  const long long aggThreads = (long long)N_EDGES * 4;
  const int aggBlocks = (int)((aggThreads + 255) / 256);
  agg_kernel<<<aggBlocks, 256, 0, stream>>>(x, ei, et, agg, cnt, 1);

  const int rBlocks = (N_NODES + R1_WAVES * 16 - 1) / (R1_WAVES * 16);
  rgcn1_kernel<<<rBlocks, R1_WAVES * 32,
                 (size_t)33 * 64 * BST * sizeof(_Float16), stream>>>(
      agg, cnt, x, w1, root1, b1, h1);

  hipMemsetAsync(agg, 0, aggBytes, stream);
  agg_kernel<<<aggBlocks, 256, 0, stream>>>(h1, ei, et, agg, cnt, 0);

  rgcn2_kernel<<<rBlocks, R1_WAVES * 32,
                 (size_t)33 * 16 * BST * sizeof(_Float16), stream>>>(
      agg, cnt, h1, w2, root2, b2, out);
}
